// ModifiedLSTMcell_31997506355812
// MI455X (gfx1250) — compile-verified, matrix operand order checked by reading, not run on hardware
//
#include <hip/hip_runtime.h>

// ---------------------------------------------------------------------------
// LSTM (torch LSTMCell semantics) for MI455X / gfx1250, bf16 WMMA pipeline.
//   Phase 0: convert x / W_ih / W_hh to bf16, bias = b_ih+b_hh, h0 -> ping[0]
//   Phase 1: x_proj[s*B+b, 4H] = x @ W_ih^T + bias      (v_wmma_f32_16x16x32_bf16)
//   Phase 2: persistent cooperative kernel, 1024 sequential steps:
//            gates = x_proj[t] + h @ W_hh^T ; cell update ; grid barrier
// ---------------------------------------------------------------------------

#define SEQ 1024
#define BB  128
#define II  512
#define HH  512
#define G4  2048        // 4*H
#define NWG 32          // workgroups in the persistent recurrence kernel

typedef __attribute__((ext_vector_type(16))) __bf16 v16bf;
typedef __attribute__((ext_vector_type(8)))  float  v8f;

union FragBF {
    v16bf v;
    uint4 q[2];
};

__device__ __forceinline__ unsigned short f2bf(float x) {
    unsigned u = __float_as_uint(x);
    unsigned r = u + 0x7FFFu + ((u >> 16) & 1u);   // round-to-nearest-even
    return (unsigned short)(r >> 16);
}
__device__ __forceinline__ float bf2f(unsigned short h) {
    return __uint_as_float(((unsigned)h) << 16);
}

// Load a 16x32 bf16 A/B fragment slice for this lane.
// CDNA5 16-bit operand layout: lane -> row (lane&15), half = lane>>4 selects
// K packing: regs 0..3 hold K = half*8 + 0..7, regs 4..7 hold K = 16+half*8+0..7.
// p must point at element (row, kstep*32 + half*8); contiguous along K.
__device__ __forceinline__ v16bf load_frag(const unsigned short* p) {
    FragBF f;
    f.q[0] = *(const uint4*)(p);        // k .. k+7
    f.q[1] = *(const uint4*)(p + 16);   // k+16 .. k+23
    return f.v;
}

__device__ __forceinline__ v8f wmma_bf16(v16bf a, v16bf b, v8f c) {
    return __builtin_amdgcn_wmma_f32_16x16x32_bf16(
        false, a, false, b, (short)0, c, false, false);
}

__device__ __forceinline__ float sigm(float x) {
    return 1.0f / (1.0f + __expf(-x));
}
__device__ __forceinline__ float tanh_f(float x) {
    // tanh(x) = 1 - 2/(exp(2x)+1)
    return 1.0f - 2.0f / (__expf(2.0f * x) + 1.0f);
}

// ------------------------------- prep kernels ------------------------------

__global__ void k_f32_to_bf16(const float* __restrict__ src,
                              unsigned short* __restrict__ dst, int n) {
    int stride = gridDim.x * blockDim.x;
    for (int i = blockIdx.x * blockDim.x + threadIdx.x; i < n; i += stride)
        dst[i] = f2bf(src[i]);
}

__global__ void k_prep(const float* __restrict__ bih, const float* __restrict__ bhh,
                       float* __restrict__ bias, const float* __restrict__ h0,
                       unsigned short* __restrict__ hping, unsigned* __restrict__ bar) {
    int tid = blockIdx.x * blockDim.x + threadIdx.x;
    if (tid < G4) bias[tid] = bih[tid] + bhh[tid];
    int stride = gridDim.x * blockDim.x;
    for (int i = tid; i < BB * HH; i += stride)
        hping[i] = f2bf(h0[i]);
    if (tid == 0) { bar[0] = 0u; bar[1] = 0u; }
}

// ------------------------- phase 1: x projection GEMM ----------------------
// out[r, n] = sum_k xb[r,k] * wih[n,k] + bias[n]   (stored bf16)
// Block: 256 threads = 8 waves arranged 2 (rows) x 4 (cols); per-wave 2x2 tiles.
// Block tile: 64 rows x 128 cols. Grid: (131072/64, 2048/128) = (2048, 16).

__global__ __launch_bounds__(256) void k_xproj(
        const unsigned short* __restrict__ xb,
        const unsigned short* __restrict__ wih,
        const float* __restrict__ bias,
        unsigned short* __restrict__ xp) {
    const int lane = threadIdx.x & 31;
    const int wave = threadIdx.x >> 5;
    const int half = lane >> 4;
    const int lr   = lane & 15;
    const int rb = blockIdx.x * 64 + (wave >> 2) * 32;   // row base (2 m-tiles)
    const int nb = blockIdx.y * 128 + (wave & 3) * 32;   // col base (2 n-tiles)

    v8f acc[2][2] = {};
    for (int ks = 0; ks < 16; ++ks) {
        const int kb = ks * 32 + half * 8;
        v16bf a0 = load_frag(xb  + (size_t)(rb + lr)      * II + kb);
        v16bf a1 = load_frag(xb  + (size_t)(rb + 16 + lr) * II + kb);
        v16bf b0 = load_frag(wih + (size_t)(nb + lr)      * II + kb);
        v16bf b1 = load_frag(wih + (size_t)(nb + 16 + lr) * II + kb);
        acc[0][0] = wmma_bf16(a0, b0, acc[0][0]);
        acc[0][1] = wmma_bf16(a0, b1, acc[0][1]);
        acc[1][0] = wmma_bf16(a1, b0, acc[1][0]);
        acc[1][1] = wmma_bf16(a1, b1, acc[1][1]);
    }
    // C/D layout: reg i -> M = i + 8*half, N = lane&15
#pragma unroll
    for (int mi = 0; mi < 2; ++mi)
#pragma unroll
        for (int ni = 0; ni < 2; ++ni) {
            const int row0 = rb + mi * 16 + half * 8;
            const int col  = nb + ni * 16 + lr;
            const float bc = bias[col];
#pragma unroll
            for (int i = 0; i < 8; ++i)
                xp[(size_t)(row0 + i) * G4 + col] = f2bf(acc[mi][ni][i] + bc);
        }
}

// ----------------------- phase 2: persistent recurrence --------------------
// 32 workgroups x 64 threads (2 waves). Workgroup wg owns hidden cols
// j0 = wg*16 .. +15 and computes gate columns {g*512 + j0 + j} for g=0..3,
// all 128 batch rows. W_hh slice lives in LDS for the whole kernel.
// Per wave: 4 m-tiles x 4 gate tiles (4x4 register blocking), K=512.

__global__ __launch_bounds__(64) void k_lstm(
        const unsigned short* __restrict__ whh,   // [2048,512] bf16
        const unsigned short* __restrict__ xp,    // [SEQ*B, 2048] bf16
        const float* __restrict__ c0,             // [B*H]
        unsigned short* __restrict__ hping,       // [2][B*H] bf16 ping-pong
        float* __restrict__ out,                  // [SEQ*B*H]
        float* __restrict__ hn_out,               // [B*H]
        float* __restrict__ cn_out,               // [B*H]
        unsigned* __restrict__ bar) {
    __shared__ unsigned short Wl[64 * 520];       // 64 W_hh rows, padded stride
    __shared__ float          gl[4 * BB * 16];    // gate accumulators (f32)
    __shared__ unsigned short xpl[4 * BB * 16];   // x_proj strips (bf16)

    const int tid  = threadIdx.x;
    const int wg   = blockIdx.x;
    const int j0   = wg * 16;
    const int lane = tid & 31;
    const int wave = tid >> 5;
    const int half = lane >> 4;
    const int lr   = lane & 15;

    // ---- preload W_hh slice: row rr = g*16 + jl  ->  W_hh[g*512 + j0 + jl, :]
    for (int idx = tid; idx < 64 * 64; idx += 64) {
        const int rr = idx >> 6;            // 0..63
        const int kc = (idx & 63) * 8;      // 8 elems = 16 B
        const int g  = rr >> 4, jl = rr & 15;
        *(uint4*)&Wl[rr * 520 + kc] =
            *(const uint4*)(whh + (size_t)(g * 512 + j0 + jl) * HH + kc);
    }

    // ---- per-thread cell state: 32 cells, idx = tid + 64*i, b = idx>>4, j = idx&15
    float c[32];
#pragma unroll
    for (int i = 0; i < 32; ++i) {
        const int idx = tid + i * 64;
        c[i] = c0[(idx >> 4) * HH + j0 + (idx & 15)];
    }
    __syncthreads();

    unsigned short* const hb0 = hping;
    unsigned short* const hb1 = hping + BB * HH;
    const int mtb = wave * 4;

    for (int t = 0; t < SEQ; ++t) {
        const unsigned short* hprev = (t & 1) ? hb1 : hb0;
        unsigned short*       hnext = (t & 1) ? hb0 : hb1;

        // ---- stage x_proj strips for this step into LDS (4 g x 128 b x 16)
        const unsigned short* xpt = xp + (size_t)t * (BB * G4);
        for (int idx = tid; idx < 4 * BB * 2; idx += 64) {
            const int g = idx >> 8;
            const int r = idx & 255;
            const int b = r >> 1, piece = (r & 1) * 8;
            *(uint4*)&xpl[g * (BB * 16) + b * 16 + piece] =
                *(const uint4*)(xpt + (size_t)b * G4 + g * 512 + j0 + piece);
        }

        // ---- recurrent GEMM: gates += h @ W_hh^T (this wg's 64 gate cols)
        v8f acc[4][4] = {};
        for (int ks = 0; ks < 16; ++ks) {
            const int kb = ks * 32 + half * 8;
            v16bf a[4], w[4];
#pragma unroll
            for (int mi = 0; mi < 4; ++mi)
                a[mi] = load_frag(hprev + (size_t)((mtb + mi) * 16 + lr) * HH + kb);
#pragma unroll
            for (int g = 0; g < 4; ++g)
                w[g] = load_frag(&Wl[(g * 16 + lr) * 520 + kb]);
#pragma unroll
            for (int mi = 0; mi < 4; ++mi)
#pragma unroll
                for (int g = 0; g < 4; ++g)
                    acc[mi][g] = wmma_bf16(a[mi], w[g], acc[mi][g]);
        }
#pragma unroll
        for (int mi = 0; mi < 4; ++mi)
#pragma unroll
            for (int g = 0; g < 4; ++g) {
                const int b0r = (mtb + mi) * 16 + half * 8;
#pragma unroll
                for (int i = 0; i < 8; ++i)
                    gl[g * (BB * 16) + (b0r + i) * 16 + lr] = acc[mi][g][i];
            }
        __syncthreads();

        // ---- cell update: each thread owns 32 (b, j) cells
        const size_t obase = (size_t)t * (BB * HH);
#pragma unroll
        for (int i = 0; i < 32; ++i) {
            const int idx = tid + i * 64;
            const int b = idx >> 4, j = idx & 15;
            const int o = b * 16 + j;
            const float gi = sigm(gl[0 * (BB * 16) + o] + bf2f(xpl[0 * (BB * 16) + o]));
            const float gf = sigm(gl[1 * (BB * 16) + o] + bf2f(xpl[1 * (BB * 16) + o]));
            const float gg = tanh_f(gl[2 * (BB * 16) + o] + bf2f(xpl[2 * (BB * 16) + o]));
            const float go = sigm(gl[3 * (BB * 16) + o] + bf2f(xpl[3 * (BB * 16) + o]));
            const float cn = gf * c[i] + gi * gg;
            const float hn = go * tanh_f(cn);
            c[i] = cn;
            out[obase + b * HH + j0 + j] = hn;
            hnext[b * HH + j0 + j] = f2bf(hn);
            if (t == SEQ - 1) hn_out[b * HH + j0 + j] = hn;
        }

        // ---- grid barrier (sense-reversing on generation counter)
        __threadfence();          // release: flush h writes toward L2
        __syncthreads();
        if (tid == 0) {
            const unsigned gen =
                __hip_atomic_load(&bar[1], __ATOMIC_RELAXED, __HIP_MEMORY_SCOPE_AGENT);
            if (__hip_atomic_fetch_add(&bar[0], 1u, __ATOMIC_ACQ_REL,
                                       __HIP_MEMORY_SCOPE_AGENT) == NWG - 1) {
                __hip_atomic_store(&bar[0], 0u, __ATOMIC_RELAXED,
                                   __HIP_MEMORY_SCOPE_AGENT);
                __hip_atomic_fetch_add(&bar[1], 1u, __ATOMIC_RELEASE,
                                       __HIP_MEMORY_SCOPE_AGENT);
            } else {
                while (__hip_atomic_load(&bar[1], __ATOMIC_ACQUIRE,
                                         __HIP_MEMORY_SCOPE_AGENT) == gen)
                    __builtin_amdgcn_s_sleep(1);
            }
        }
        __syncthreads();
        __threadfence();          // acquire: invalidate stale WGP$ lines of h
    }

    // ---- final cell state
#pragma unroll
    for (int i = 0; i < 32; ++i) {
        const int idx = tid + i * 64;
        cn_out[(idx >> 4) * HH + j0 + (idx & 15)] = c[i];
    }
}

// --------------------------------- host ------------------------------------

extern "C" void kernel_launch(void* const* d_in, const int* in_sizes, int n_in,
                              void* d_out, int out_size, void* d_ws, size_t ws_size,
                              hipStream_t stream) {
    (void)in_sizes; (void)n_in; (void)out_size; (void)ws_size;
    const float* x   = (const float*)d_in[0];
    const float* h0  = (const float*)d_in[1];
    const float* c0  = (const float*)d_in[2];
    const float* Wih = (const float*)d_in[3];
    const float* Whh = (const float*)d_in[4];
    const float* bih = (const float*)d_in[5];
    const float* bhh = (const float*)d_in[6];
    float* out = (float*)d_out;

    char* ws = (char*)d_ws;
    size_t off = 0;
    auto carve = [&](size_t bytes) -> void* {
        void* p = ws + off;
        off = (off + bytes + 255) & ~(size_t)255;
        return p;
    };
    unsigned short* xb    = (unsigned short*)carve((size_t)SEQ * BB * II * 2);
    unsigned short* wihb  = (unsigned short*)carve((size_t)G4 * II * 2);
    unsigned short* whhb  = (unsigned short*)carve((size_t)G4 * HH * 2);
    float*          bias  = (float*)carve((size_t)G4 * 4);
    unsigned short* hping = (unsigned short*)carve((size_t)2 * BB * HH * 2);
    unsigned*       bar   = (unsigned*)carve(256);
    unsigned short* xp    = (unsigned short*)carve((size_t)SEQ * BB * G4 * 2);

    k_f32_to_bf16<<<4096, 256, 0, stream>>>(x, xb, SEQ * BB * II);
    k_f32_to_bf16<<<1024, 256, 0, stream>>>(Wih, wihb, G4 * II);
    k_f32_to_bf16<<<1024, 256, 0, stream>>>(Whh, whhb, G4 * HH);
    k_prep<<<256, 256, 0, stream>>>(bih, bhh, bias, h0, hping, bar);

    k_xproj<<<dim3((SEQ * BB) / 64, G4 / 128), 256, 0, stream>>>(xb, wihb, bias, xp);

    float* hn_out = out + (size_t)SEQ * BB * HH;
    float* cn_out = hn_out + (size_t)BB * HH;
    k_lstm<<<NWG, 64, 0, stream>>>(whhb, xp, c0, hping, out, hn_out, cn_out, bar);
}
